// pmfDiT_12214886990804
// MI455X (gfx1250) — matrix operand order repaired and checked by reading, not compiled, not run on hardware
//
#include <hip/hip_runtime.h>
#include <hip/hip_bf16.h>

typedef __attribute__((ext_vector_type(16))) _Float16 v16h;
typedef __attribute__((ext_vector_type(8)))  _Float16 v8h;
typedef __attribute__((ext_vector_type(8)))  float    v8f;

#define D_MODEL  768
#define N_HEADS  12
#define HDIM     64
#define SEQ_LEN  276
#define SEQ_PAD  288
#define BATCH_N  32
#define MLP_H    2048
#define PREFIX_N 20
#define TOKENS   (BATCH_N * SEQ_PAD)   /* 9216 */
#define QKV_W    (3 * D_MODEL)         /* 2304 fused QKV output width */
#define G13_W    (2 * MLP_H)           /* 4096 fused w1/w3 output width */
#define EPS_F    1e-6f

static __device__ __forceinline__ v16h pack16(v8h lo, v8h hi) {
  v16h r;
#pragma unroll
  for (int e = 0; e < 8; ++e) { r[e] = lo[e]; r[e + 8] = hi[e]; }
  return r;
}

// ---------------------------------------------------------------------------
// Weight convert + transpose: W (K x N, f32 row-major) -> Wt (N x K, f16)
// ---------------------------------------------------------------------------
__global__ void cvt_transpose_kernel(const float* __restrict__ W,
                                     _Float16* __restrict__ Wt, int K, int N) {
  int i = blockIdx.x * blockDim.x + threadIdx.x;
  if (i >= K * N) return;
  int k = i / N, n = i % N;
  Wt[(size_t)n * K + k] = (_Float16)W[i];
}

// ---------------------------------------------------------------------------
// RMSNorm over D_MODEL, write f16 (padded tokens -> zeros). One block / token.
// ---------------------------------------------------------------------------
__global__ void rmsnorm1_kernel(const float* __restrict__ x,
                                const float* __restrict__ w,
                                _Float16* __restrict__ h) {
  int tok = blockIdx.x;
  int b = tok / SEQ_PAD, s = tok % SEQ_PAD;
  int t = threadIdx.x;
  size_t row = (size_t)tok * D_MODEL;
  if (s >= SEQ_LEN) {
    for (int i = t; i < D_MODEL; i += 256) h[row + i] = (_Float16)0.f;
    return;
  }
  size_t xrow = ((size_t)b * SEQ_LEN + s) * D_MODEL;
  float vals[3]; float ss = 0.f;
#pragma unroll
  for (int j = 0; j < 3; ++j) { int i = t + j * 256; float v = x[xrow + i]; vals[j] = v; ss += v * v; }
  __shared__ float red[8];
  for (int off = 16; off >= 1; off >>= 1) ss += __shfl_xor(ss, off, 32);
  if ((t & 31) == 0) red[t >> 5] = ss;
  __syncthreads();
  if (t < 8) {
    float v = red[t];
    for (int off = 4; off >= 1; off >>= 1) v += __shfl_xor(v, off, 32);
    if (t == 0) red[0] = v;
  }
  __syncthreads();
  float r = rsqrtf(red[0] * (1.0f / D_MODEL) + EPS_F);
#pragma unroll
  for (int j = 0; j < 3; ++j) { int i = t + j * 256; h[row + i] = (_Float16)(vals[j] * r * w[i]); }
}

// ---------------------------------------------------------------------------
// GEMM: C[M x N] f32 = A[M x K] f16 (row major) * Bt[N x K] f16 (B transposed)
// 64x64 wave tile (4x4 WMMA accumulators, 32 FLOP/byte from L2).
// K/N are template constants so every fragment address folds into the 24-bit
// global_load immediate off a single base pointer (no per-row pointers, no
// spills). launch_bounds(256,1) gives the allocator the full 256-VGPR file.
// M % 64 == 0.
// ---------------------------------------------------------------------------
template <int K, int N>
__global__ __launch_bounds__(256, 1)
void gemm_f16f32(const _Float16* __restrict__ A,
                 const _Float16* __restrict__ Bt,
                 float* __restrict__ C, int M) {
  int gid  = blockIdx.x * blockDim.x + threadIdx.x;
  int wave = gid >> 5;
  int lane = gid & 31;
  constexpr int NT = N / 64;
  int mt = wave / NT;
  int nt = wave - mt * NT;
  if (mt * 64 >= M) return;
  int lm = lane & 15;
  int hi = lane >> 4;

  const _Float16* Abase = A  + (size_t)(mt * 64 + lm) * K + hi * 16;
  const _Float16* Bbase = Bt + (size_t)(nt * 64 + lm) * K + hi * 16;

  v8f zero = {};
  v8f acc[4][4];
#pragma unroll
  for (int i = 0; i < 4; ++i)
#pragma unroll
    for (int j = 0; j < 4; ++j) acc[i][j] = zero;

  for (int k0 = 0; k0 < K; k0 += 32) {
    v16h af[4];
#pragma unroll
    for (int i = 0; i < 4; ++i)
      af[i] = pack16(*(const v8h*)(Abase + i * 16 * K + k0),
                     *(const v8h*)(Abase + i * 16 * K + k0 + 8));
#pragma unroll
    for (int j = 0; j < 4; ++j) {
      v16h bf = pack16(*(const v8h*)(Bbase + j * 16 * K + k0),
                       *(const v8h*)(Bbase + j * 16 * K + k0 + 8));
#pragma unroll
      for (int i = 0; i < 4; ++i)
        acc[i][j] = __builtin_amdgcn_wmma_f32_16x16x32_f16(false, af[i], false, bf,
                                                           (short)0, acc[i][j], false, false);
    }
  }

#pragma unroll
  for (int i = 0; i < 4; ++i) {
    float* Crow = C + (size_t)(mt * 64 + i * 16 + 8 * hi) * N + nt * 64 + lm;
#pragma unroll
    for (int j = 0; j < 4; ++j)
#pragma unroll
      for (int r = 0; r < 8; ++r)
        Crow[(size_t)r * N + j * 16] = acc[i][j][r];
  }
}

// ---------------------------------------------------------------------------
// Per-head RMSNorm + RoPE + 1/sqrt(Dh) scale on the fused QKV output
// (row stride QKV_W; q at col h*64, k at 768+h*64, v at 1536+h*64).
// Repack q/k as [b,h,s,d] f16 and v transposed as [b,h,d,s] f16.
// One wave per (b,h,s); lane owns dims 2l, 2l+1.
// ---------------------------------------------------------------------------
__global__ void qk_post_kernel(const float* __restrict__ qkv,
                               const float* __restrict__ qw, const float* __restrict__ kw,
                               const float* __restrict__ cosT, const float* __restrict__ sinT,
                               _Float16* __restrict__ q16, _Float16* __restrict__ k16,
                               _Float16* __restrict__ v16t) {
  int gid = blockIdx.x * blockDim.x + threadIdx.x;
  int wave = gid >> 5, lane = gid & 31;
  int s = wave % SEQ_PAD;
  int h = (wave / SEQ_PAD) % N_HEADS;
  int b = wave / (SEQ_PAD * N_HEADS);
  if (b >= BATCH_N) return;
  size_t qkrow  = ((size_t)(b * N_HEADS + h) * SEQ_PAD + s) * HDIM;
  size_t vtbase = (size_t)(b * N_HEADS + h) * HDIM * SEQ_PAD;
  int de = 2 * lane, dod = 2 * lane + 1;
  if (s >= SEQ_LEN) {
    q16[qkrow + de] = (_Float16)0.f; q16[qkrow + dod] = (_Float16)0.f;
    k16[qkrow + de] = (_Float16)0.f; k16[qkrow + dod] = (_Float16)0.f;
    v16t[vtbase + (size_t)de  * SEQ_PAD + s] = (_Float16)0.f;
    v16t[vtbase + (size_t)dod * SEQ_PAD + s] = (_Float16)0.f;
    return;
  }
  size_t src = (size_t)(b * SEQ_PAD + s) * QKV_W + h * HDIM;
  float qe = qkv[src + de],           qo = qkv[src + dod];
  float ke = qkv[src + D_MODEL + de], ko = qkv[src + D_MODEL + dod];
  float ssq = qe * qe + qo * qo;
  float ssk = ke * ke + ko * ko;
  for (int off = 16; off >= 1; off >>= 1) {
    ssq += __shfl_xor(ssq, off, 32);
    ssk += __shfl_xor(ssk, off, 32);
  }
  float rq = rsqrtf(ssq * (1.0f / HDIM) + EPS_F);
  float rk = rsqrtf(ssk * (1.0f / HDIM) + EPS_F);
  qe *= rq * qw[de]; qo *= rq * qw[dod];
  ke *= rk * kw[de]; ko *= rk * kw[dod];
  if (s >= PREFIX_N) {
    int p = s - PREFIX_N;
    float c = cosT[p * 32 + lane], sn = sinT[p * 32 + lane];
    float t;
    t = qe * c - qo * sn; qo = qe * sn + qo * c; qe = t;
    t = ke * c - ko * sn; ko = ke * sn + ko * c; ke = t;
  }
  qe *= 0.125f; qo *= 0.125f;   // 1/sqrt(64)
  q16[qkrow + de] = (_Float16)qe; q16[qkrow + dod] = (_Float16)qo;
  k16[qkrow + de] = (_Float16)ke; k16[qkrow + dod] = (_Float16)ko;
  v16t[vtbase + (size_t)de  * SEQ_PAD + s] = (_Float16)qkv[src + 2 * D_MODEL + de];
  v16t[vtbase + (size_t)dod * SEQ_PAD + s] = (_Float16)qkv[src + 2 * D_MODEL + dod];
}

// ---------------------------------------------------------------------------
// Flash-style attention: one wave per (b, h, 16-row q tile). 32 keys / iter.
// QK^T and P*V on WMMA; online softmax via half-wave shuffles; P transposed
// through LDS into an A-fragment. Output packed token-major f16 for Wo GEMM.
// launch_bounds(32,1): single-wave workgroup gets the full VGPR file so the
// Q fragments / accumulators / K,V pointers stay resident (no scratch spills).
// ---------------------------------------------------------------------------
__global__ __launch_bounds__(32, 1)
void attention_kernel(const _Float16* __restrict__ q16,
                      const _Float16* __restrict__ k16,
                      const _Float16* __restrict__ v16t,
                      _Float16* __restrict__ attn16) {
  __shared__ _Float16 pls[16 * 32];
  int wid  = blockIdx.x;
  int lane = threadIdx.x & 31;
  int qt = wid % (SEQ_PAD / 16);
  int h  = (wid / (SEQ_PAD / 16)) % N_HEADS;
  int b  = wid / ((SEQ_PAD / 16) * N_HEADS);
  int lm = lane & 15, hi = lane >> 4;

  const _Float16* Q  = q16  + ((size_t)(b * N_HEADS + h) * SEQ_PAD + qt * 16) * HDIM;
  const _Float16* Kb = k16  + (size_t)(b * N_HEADS + h) * SEQ_PAD * HDIM;
  const _Float16* Vt = v16t + (size_t)(b * N_HEADS + h) * HDIM * SEQ_PAD;

  v16h qa0 = pack16(*(const v8h*)(Q + lm * HDIM + hi * 16),
                    *(const v8h*)(Q + lm * HDIM + hi * 16 + 8));
  v16h qa1 = pack16(*(const v8h*)(Q + lm * HDIM + 32 + hi * 16),
                    *(const v8h*)(Q + lm * HDIM + 32 + hi * 16 + 8));

  float mrow[8], lrow[8];
#pragma unroll
  for (int r = 0; r < 8; ++r) { mrow[r] = -3.0e38f; lrow[r] = 0.f; }
  v8f acc0 = {}, acc1 = {}, acc2 = {}, acc3 = {};

  for (int kb = 0; kb < SEQ_PAD; kb += 32) {
    // ---- scores for 32 keys (two 16-wide tiles) ----
    const _Float16* K0 = Kb + (size_t)(kb + lm) * HDIM;
    const _Float16* K1 = Kb + (size_t)(kb + 16 + lm) * HDIM;
    v8f s0 = {}, s1 = {};
    s0 = __builtin_amdgcn_wmma_f32_16x16x32_f16(false, qa0, false,
          pack16(*(const v8h*)(K0 + hi * 16), *(const v8h*)(K0 + hi * 16 + 8)),
          (short)0, s0, false, false);
    s0 = __builtin_amdgcn_wmma_f32_16x16x32_f16(false, qa1, false,
          pack16(*(const v8h*)(K0 + 32 + hi * 16), *(const v8h*)(K0 + 32 + hi * 16 + 8)),
          (short)0, s0, false, false);
    s1 = __builtin_amdgcn_wmma_f32_16x16x32_f16(false, qa0, false,
          pack16(*(const v8h*)(K1 + hi * 16), *(const v8h*)(K1 + hi * 16 + 8)),
          (short)0, s1, false, false);
    s1 = __builtin_amdgcn_wmma_f32_16x16x32_f16(false, qa1, false,
          pack16(*(const v8h*)(K1 + 32 + hi * 16), *(const v8h*)(K1 + 32 + hi * 16 + 8)),
          (short)0, s1, false, false);

    bool msk0 = (kb + lm)      >= SEQ_LEN;
    bool msk1 = (kb + 16 + lm) >= SEQ_LEN;

    // ---- online softmax (row = r + 8*hi lives on a 16-lane group) ----
#pragma unroll
    for (int r = 0; r < 8; ++r) {
      float a = msk0 ? -1.0e30f : s0[r];
      float c = msk1 ? -1.0e30f : s1[r];
      float v = fmaxf(a, c);
#pragma unroll
      for (int off = 8; off >= 1; off >>= 1) v = fmaxf(v, __shfl_xor(v, off, 32));
      float mn = fmaxf(mrow[r], v);
      float sc = __expf(mrow[r] - mn);
      float p0 = __expf(a - mn);
      float p1 = __expf(c - mn);
      float ps = p0 + p1;
#pragma unroll
      for (int off = 8; off >= 1; off >>= 1) ps += __shfl_xor(ps, off, 32);
      mrow[r] = mn;
      lrow[r] = lrow[r] * sc + ps;
      acc0[r] *= sc; acc1[r] *= sc; acc2[r] *= sc; acc3[r] *= sc;
      pls[(r + 8 * hi) * 32 + lm]      = (_Float16)p0;
      pls[(r + 8 * hi) * 32 + 16 + lm] = (_Float16)p1;
    }
    __syncthreads();
    // transpose P (C-layout) into an A-fragment (16x32, K = key index)
    v16h pa = pack16(*(const v8h*)(pls + lm * 32 + hi * 16),
                     *(const v8h*)(pls + lm * 32 + hi * 16 + 8));
    __syncthreads();

    // ---- P * V over this 32-key chunk; four d-tiles of 16 ----
    const _Float16* V0 = Vt + (size_t)(lm)      * SEQ_PAD + kb + hi * 16;
    const _Float16* V1 = Vt + (size_t)(16 + lm) * SEQ_PAD + kb + hi * 16;
    const _Float16* V2 = Vt + (size_t)(32 + lm) * SEQ_PAD + kb + hi * 16;
    const _Float16* V3 = Vt + (size_t)(48 + lm) * SEQ_PAD + kb + hi * 16;
    acc0 = __builtin_amdgcn_wmma_f32_16x16x32_f16(false, pa, false,
            pack16(*(const v8h*)V0, *(const v8h*)(V0 + 8)), (short)0, acc0, false, false);
    acc1 = __builtin_amdgcn_wmma_f32_16x16x32_f16(false, pa, false,
            pack16(*(const v8h*)V1, *(const v8h*)(V1 + 8)), (short)0, acc1, false, false);
    acc2 = __builtin_amdgcn_wmma_f32_16x16x32_f16(false, pa, false,
            pack16(*(const v8h*)V2, *(const v8h*)(V2 + 8)), (short)0, acc2, false, false);
    acc3 = __builtin_amdgcn_wmma_f32_16x16x32_f16(false, pa, false,
            pack16(*(const v8h*)V3, *(const v8h*)(V3 + 8)), (short)0, acc3, false, false);
  }

  // ---- finalize: divide by l, store token-major f16 for the Wo GEMM ----
  _Float16* Obase = attn16 + ((size_t)b * SEQ_PAD + qt * 16 + 8 * hi) * D_MODEL
                           + h * HDIM + lm;
#pragma unroll
  for (int r = 0; r < 8; ++r) {
    float inv = 1.0f / lrow[r];
    _Float16* Or = Obase + (size_t)r * D_MODEL;
    Or[0]  = (_Float16)(acc0[r] * inv);
    Or[16] = (_Float16)(acc1[r] * inv);
    Or[32] = (_Float16)(acc2[r] * inv);
    Or[48] = (_Float16)(acc3[r] * inv);
  }
}

// ---------------------------------------------------------------------------
// x2 = x + attn_out * attn_scale ; h2 = rmsnorm(x2) * norm2_w (f16)
// ---------------------------------------------------------------------------
__global__ void resid_norm_kernel(const float* __restrict__ x, const float* __restrict__ of,
                                  const float* __restrict__ ascale, const float* __restrict__ n2w,
                                  float* __restrict__ x2, _Float16* __restrict__ h2) {
  int tok = blockIdx.x;
  int b = tok / SEQ_PAD, s = tok % SEQ_PAD;
  int t = threadIdx.x;
  size_t row = (size_t)tok * D_MODEL;
  if (s >= SEQ_LEN) {
    for (int i = t; i < D_MODEL; i += 256) { x2[row + i] = 0.f; h2[row + i] = (_Float16)0.f; }
    return;
  }
  size_t xrow = ((size_t)b * SEQ_LEN + s) * D_MODEL;
  float vals[3]; float ss = 0.f;
#pragma unroll
  for (int j = 0; j < 3; ++j) {
    int i = t + j * 256;
    float v = x[xrow + i] + of[row + i] * ascale[i];
    vals[j] = v; x2[row + i] = v; ss += v * v;
  }
  __shared__ float red[8];
  for (int off = 16; off >= 1; off >>= 1) ss += __shfl_xor(ss, off, 32);
  if ((t & 31) == 0) red[t >> 5] = ss;
  __syncthreads();
  if (t < 8) {
    float v = red[t];
    for (int off = 4; off >= 1; off >>= 1) v += __shfl_xor(v, off, 32);
    if (t == 0) red[0] = v;
  }
  __syncthreads();
  float r = rsqrtf(red[0] * (1.0f / D_MODEL) + EPS_F);
#pragma unroll
  for (int j = 0; j < 3; ++j) { int i = t + j * 256; h2[row + i] = (_Float16)(vals[j] * r * n2w[i]); }
}

// ---------------------------------------------------------------------------
// silu(g1) * g3 -> f16   (fused g13 buffer: row stride G13_W, g3 at +MLP_H)
// ---------------------------------------------------------------------------
__global__ void swiglu_kernel(const float* __restrict__ g13,
                              _Float16* __restrict__ m16, int n) {
  int i = blockIdx.x * blockDim.x + threadIdx.x;
  if (i >= n) return;
  int row = i / MLP_H, col = i - row * MLP_H;
  size_t base = (size_t)row * G13_W + col;
  float g = g13[base];
  float si = g / (1.0f + __expf(-g));
  m16[i] = (_Float16)(si * g13[base + MLP_H]);
}

// ---------------------------------------------------------------------------
// out = x2 + mlp_out * mlp_scale   (strip seq padding)
// ---------------------------------------------------------------------------
__global__ void final_kernel(const float* __restrict__ x2, const float* __restrict__ mo,
                             const float* __restrict__ mscale, float* __restrict__ out) {
  int i = blockIdx.x * blockDim.x + threadIdx.x;
  const int n = BATCH_N * SEQ_LEN * D_MODEL;
  if (i >= n) return;
  int d = i % D_MODEL;
  int s = (i / D_MODEL) % SEQ_LEN;
  int b = i / (D_MODEL * SEQ_LEN);
  size_t tok = (size_t)b * SEQ_PAD + s;
  out[i] = x2[tok * D_MODEL + d] + mo[tok * D_MODEL + d] * mscale[d];
}

// ===========================================================================
extern "C" void kernel_launch(void* const* d_in, const int* in_sizes, int n_in,
                              void* d_out, int out_size, void* d_ws, size_t ws_size,
                              hipStream_t stream) {
  (void)in_sizes; (void)n_in; (void)out_size; (void)ws_size;
  const float* x        = (const float*)d_in[0];
  const float* Wq       = (const float*)d_in[1];
  const float* Wk       = (const float*)d_in[2];
  const float* Wv       = (const float*)d_in[3];
  const float* Wo       = (const float*)d_in[4];
  const float* q_norm_w = (const float*)d_in[5];
  const float* k_norm_w = (const float*)d_in[6];
  const float* norm1_w  = (const float*)d_in[7];
  const float* norm2_w  = (const float*)d_in[8];
  const float* attn_sc  = (const float*)d_in[9];
  const float* mlp_sc   = (const float*)d_in[10];
  const float* w1       = (const float*)d_in[11];
  const float* w2       = (const float*)d_in[12];
  const float* w3       = (const float*)d_in[13];
  const float* rope_cos = (const float*)d_in[14];
  const float* rope_sin = (const float*)d_in[15];

  const size_t DD = (size_t)D_MODEL * D_MODEL;
  const size_t DM = (size_t)D_MODEL * MLP_H;
  const size_t TD = (size_t)TOKENS * D_MODEL;
  const size_t TM = (size_t)TOKENS * MLP_H;
  const size_t TQ = (size_t)TOKENS * QKV_W;   // fused QKV activations
  const size_t TG = (size_t)TOKENS * G13_W;   // fused w1/w3 activations

  char* ws = (char*)d_ws;
  size_t off = 0;
  auto alloc = [&](size_t bytes) -> char* {
    char* p = ws + off;
    off += (bytes + 255) & ~(size_t)255;
    return p;
  };

  _Float16* qkvT = (_Float16*)alloc(3 * DD * 2);       // [2304 x 768] = WqT|WkT|WvT
  _Float16* woT  = (_Float16*)alloc(DD * 2);
  _Float16* w13T = (_Float16*)alloc(2 * DM * 2);       // [4096 x 768] = w1T|w3T
  _Float16* w2T  = (_Float16*)alloc(DM * 2);           // [768 x 2048]
  _Float16* h16  = (_Float16*)alloc(TD * 2);           // reused as h2
  size_t bigF_elems = (TQ > TG) ? TQ : TG;
  float* bigF = (float*)alloc(bigF_elems * 4);         // qkv f32 -> later g13
  float* qkvF = bigF;
  float* g13  = bigF;
  size_t bigH_elems = (3 * TD > TM) ? 3 * TD : TM;
  _Float16* bigH = (_Float16*)alloc(bigH_elems * 2);   // q16/k16/vT -> later m16
  _Float16* q16 = bigH; _Float16* k16 = bigH + TD; _Float16* v16t = bigH + 2 * TD;
  _Float16* m16 = bigH;
  _Float16* attn16 = (_Float16*)alloc(TD * 2);
  float* oF = (float*)alloc(TD * 4);                   // attn@Wo -> later mlp out
  float* x2 = (float*)alloc(TD * 4);

  // weights -> f16, transposed to N x K (QKV and w1/w3 into fused panels)
  cvt_transpose_kernel<<<(int)((DD + 255) / 256), 256, 0, stream>>>(Wq, qkvT,          D_MODEL, D_MODEL);
  cvt_transpose_kernel<<<(int)((DD + 255) / 256), 256, 0, stream>>>(Wk, qkvT + DD,     D_MODEL, D_MODEL);
  cvt_transpose_kernel<<<(int)((DD + 255) / 256), 256, 0, stream>>>(Wv, qkvT + 2 * DD, D_MODEL, D_MODEL);
  cvt_transpose_kernel<<<(int)((DD + 255) / 256), 256, 0, stream>>>(Wo, woT,           D_MODEL, D_MODEL);
  cvt_transpose_kernel<<<(int)((DM + 255) / 256), 256, 0, stream>>>(w1, w13T,          D_MODEL, MLP_H);
  cvt_transpose_kernel<<<(int)((DM + 255) / 256), 256, 0, stream>>>(w3, w13T + DM,     D_MODEL, MLP_H);
  cvt_transpose_kernel<<<(int)((DM + 255) / 256), 256, 0, stream>>>(w2, w2T,           MLP_H, D_MODEL);

  // norm1 -> h (f16, padded)
  rmsnorm1_kernel<<<TOKENS, 256, 0, stream>>>(x, norm1_w, h16);

  // fused QKV projection: [9216 x 768] x [768 x 2304]
  {
    int blocks = ((TOKENS / 64) * (QKV_W / 64) + 7) / 8;
    gemm_f16f32<D_MODEL, QKV_W><<<blocks, 256, 0, stream>>>(h16, qkvT, qkvF, TOKENS);
  }

  // per-head norm + RoPE + repack
  int qkwaves = BATCH_N * N_HEADS * SEQ_PAD;
  qk_post_kernel<<<(qkwaves * 32 + 255) / 256, 256, 0, stream>>>(
      qkvF, q_norm_w, k_norm_w, rope_cos, rope_sin, q16, k16, v16t);

  // attention
  attention_kernel<<<BATCH_N * N_HEADS * (SEQ_PAD / 16), 32, 0, stream>>>(
      q16, k16, v16t, attn16);

  // output projection: [9216 x 768] x [768 x 768]
  {
    int blocks = ((TOKENS / 64) * (D_MODEL / 64) + 7) / 8;
    gemm_f16f32<D_MODEL, D_MODEL><<<blocks, 256, 0, stream>>>(attn16, woT, oF, TOKENS);
  }

  // residual + norm2
  resid_norm_kernel<<<TOKENS, 256, 0, stream>>>(x, oF, attn_sc, norm2_w, x2, h16);

  // fused MLP up: [9216 x 768] x [768 x 4096]
  {
    int blocks = ((TOKENS / 64) * (G13_W / 64) + 7) / 8;
    gemm_f16f32<D_MODEL, G13_W><<<blocks, 256, 0, stream>>>(h16, w13T, g13, TOKENS);
  }
  swiglu_kernel<<<(int)((TM + 255) / 256), 256, 0, stream>>>(g13, m16, (int)TM);
  // MLP down: [9216 x 2048] x [2048 x 768]
  {
    int blocks = ((TOKENS / 64) * (D_MODEL / 64) + 7) / 8;
    gemm_f16f32<MLP_H, D_MODEL><<<blocks, 256, 0, stream>>>(m16, w2T, oF, TOKENS);
  }

  // final residual (strip padding)
  final_kernel<<<(BATCH_N * SEQ_LEN * D_MODEL + 255) / 256, 256, 0, stream>>>(
      x2, oF, mlp_sc, (float*)d_out);
}
